// FastChannelAttention_56607668962024
// MI455X (gfx1250) — compile-verified, matrix operand order checked by reading, not run on hardware
//
#include <hip/hip_runtime.h>
#include <stdint.h>

// FastChannelAttention on MI455X (gfx1250, wave32).
// [K0] one-shot fp32->bf16 conversion of x / Wqkv / Wproj (memory bound)
// [K1] qkv = x@Wqkv^T        (bf16 WMMA, f32 acc, software-pipelined loads)
// [K2] per (b,h): G = Q^T K, L2 norms, softmax -> attn (bf16), LDS-transposed staging
// [K3] out = attn @ V^T      (TDM tensor_load_to_lds stages V + attn tiles into LDS)
// [K4] final = out@Wproj^T + bproj (f32 out, software-pipelined)

typedef __attribute__((ext_vector_type(16))) __bf16 v16bf;
typedef __attribute__((ext_vector_type(8)))  float  v8f;
typedef __attribute__((ext_vector_type(8)))  unsigned short u16x8;
typedef __attribute__((ext_vector_type(4)))  uint32_t u32x4;
typedef __attribute__((ext_vector_type(8)))  uint32_t u32x8;

union BfVec { v16bf v; u16x8 h[2]; uint16_t u[16]; };
union U8x16 { u16x8 v; uint16_t u[8]; };

__device__ __forceinline__ uint16_t f2bf(float f) {
    union { float f; uint32_t u; } c; c.f = f;
    uint32_t r = c.u + 0x7FFFu + ((c.u >> 16) & 1u);
    return (uint16_t)(r >> 16);
}
__device__ __forceinline__ float bf2f(uint16_t b) {
    union { uint32_t u; float f; } c; c.u = ((uint32_t)b) << 16;
    return c.f;
}

#define WMMA_BF16(A, B, C) \
    __builtin_amdgcn_wmma_f32_16x16x32_bf16(false, (A), false, (B), (short)0, (C), false, false)

// load 16 bf16 lane-fragment: 8 contiguous @p, 8 contiguous @p+16 (16B aligned)
__device__ __forceinline__ void load_frag(BfVec& v, const uint16_t* p) {
    v.h[0] = *(const u16x8*)(p);
    v.h[1] = *(const u16x8*)(p + 16);
}

// ------------------------------------------------------------
// TDM: DMA a 2-D tile (row-major, 2-byte elements) from global to LDS.
// D# per CDNA5 ISA ch.8: group0 = {count, lds_addr, global_addr, type=2},
// group1 = {data_size=1(2B), tensor dims, tile dims, dim0 stride}.
// 2-SGPR-group form (<=2D tensor): VADDR2/VADDR3 = NULL.
// ------------------------------------------------------------
__device__ __forceinline__ void tdm_load_tile_2d(
    uint32_t lds_byte_off, const uint16_t* gsrc,
    uint32_t tile_d0, uint32_t tile_d1,
    uint32_t tensor_d0, uint32_t tensor_d1,
    uint64_t stride0_elems)
{
    const uint64_t ga = (uint64_t)(uintptr_t)gsrc;
    u32x4 g0;
    g0[0] = 1u;                                        // count=1 (valid user D#)
    g0[1] = lds_byte_off;                              // lds_addr [63:32]
    g0[2] = (uint32_t)ga;                              // global_addr[31:0]
    g0[3] = ((uint32_t)(ga >> 32) & 0x01FFFFFFu)       // global_addr[56:32]
          | (2u << 30);                                // type=2 ("image")
    u32x8 g1;
    g1[0] = (1u << 16);                                // wg_mask=0, data_size=1 (2B)
    g1[1] = (tensor_d0 & 0xFFFFu) << 16;               // tensor_dim0[15:0] @bit48
    g1[2] = (tensor_d0 >> 16) | ((tensor_d1 & 0xFFFFu) << 16);
    g1[3] = (tensor_d1 >> 16) | (tile_d0 << 16);       // tile_dim0 @bit112
    g1[4] = tile_d1;                                   // tile_dim1 (tile_dim2=0)
    g1[5] = (uint32_t)stride0_elems;                   // tensor_dim0_stride[31:0]
    g1[6] = (uint32_t)(stride0_elems >> 32) & 0xFFFFu; // stride[47:32] | dim1_stride[15:0]=0
    g1[7] = 0u;                                        // dim1_stride[47:16]=0
    asm volatile("tensor_load_to_lds %0, %1" :: "s"(g0), "s"(g1) : "memory");
}

// ============================================================
// K0: elementwise fp32 -> bf16 (8 elements / thread)
// ============================================================
__global__ __launch_bounds__(256) void k0_cvt(const float* __restrict__ src,
                                              uint16_t* __restrict__ dst, int n8)
{
    const int i = blockIdx.x * 256 + threadIdx.x;
    if (i >= n8) return;
    const float4 f0 = *(const float4*)(src + (size_t)i * 8);
    const float4 f1 = *(const float4*)(src + (size_t)i * 8 + 4);
    U8x16 o;
    o.u[0] = f2bf(f0.x); o.u[1] = f2bf(f0.y); o.u[2] = f2bf(f0.z); o.u[3] = f2bf(f0.w);
    o.u[4] = f2bf(f1.x); o.u[5] = f2bf(f1.y); o.u[6] = f2bf(f1.z); o.u[7] = f2bf(f1.w);
    *(u16x8*)(dst + (size_t)i * 8) = o.v;
}

// ============================================================
// K1: qkv[m,d] = sum_c x[m,c]*Wqkv[d,c]; 2048 mt x 24 dg waves,
// 16x64 strip per wave, register double-buffered (hide load latency)
// ============================================================
__global__ __launch_bounds__(256) void k1_qkv_gemm(
    const uint16_t* __restrict__ xb, const uint16_t* __restrict__ wqb,
    uint16_t* __restrict__ qkv)
{
    const int lane = threadIdx.x & 31;
    const int gw   = blockIdx.x * 8 + (threadIdx.x >> 5);
    const int mt   = gw / 24;
    const int dg   = gw % 24;
    const int row  = lane & 15;
    const int half = lane >> 4;

    const uint16_t* arow = xb + (size_t)(mt * 16 + row) * 512 + half * 8;
    const uint16_t* brow[4];
#pragma unroll
    for (int j = 0; j < 4; ++j)
        brow[j] = wqb + (size_t)(dg * 64 + j * 16 + row) * 512 + half * 8;

    v8f acc[4] = {};
    BfVec a0, a1, b0[4], b1[4];

    auto ldgrp = [&](BfVec& a, BfVec (&b)[4], int kk) {
        load_frag(a, arow + kk);
#pragma unroll
        for (int j = 0; j < 4; ++j) load_frag(b[j], brow[j] + kk);
    };
    auto mmgrp = [&](BfVec& a, BfVec (&b)[4]) {
#pragma unroll
        for (int j = 0; j < 4; ++j) acc[j] = WMMA_BF16(a.v, b[j].v, acc[j]);
    };

    ldgrp(a0, b0, 0);
#pragma unroll
    for (int kk = 0; kk < 512; kk += 64) {
        if (kk + 32 < 512) ldgrp(a1, b1, kk + 32);
        mmgrp(a0, b0);
        if (kk + 64 < 512) ldgrp(a0, b0, kk + 64);
        mmgrp(a1, b1);
    }
#pragma unroll
    for (int j = 0; j < 4; ++j)
#pragma unroll
        for (int i = 0; i < 8; ++i) {
            const int m = mt * 16 + half * 8 + i;
            qkv[(size_t)m * 1536 + dg * 64 + j * 16 + row] = f2bf(acc[j][i]);
        }
}

// ============================================================
// K2: per (b,h): G[c,d]=sum_n q[n,c]k[n,d]; L2-normalize, softmax -> attn
// ============================================================
__global__ __launch_bounds__(256) void k2_chan_attn(
    const uint16_t* __restrict__ qkv, const float* __restrict__ temp,
    uint16_t* __restrict__ attn_out)
{
    __shared__ uint16_t qT[64][40];
    __shared__ uint16_t kT[64][40];
    __shared__ float ssq[64], ssk[64];
    __shared__ float attn[64][65];

    const int bh = blockIdx.x;
    const int b = bh >> 3, h = bh & 7;
    const int tid  = threadIdx.x;
    const int lane = tid & 31;
    const int wave = tid >> 5;
    const int row  = lane & 15;
    const int half = lane >> 4;

    if (tid < 64) { ssq[tid] = 0.f; ssk[tid] = 0.f; }
    __syncthreads();

    const int nl = tid >> 3;
    const int c0 = (tid & 7) * 8;
    float pq[8] = {0,0,0,0,0,0,0,0};
    float pk[8] = {0,0,0,0,0,0,0,0};

    const int t0 = wave * 2, t1 = wave * 2 + 1;
    v8f acc0 = {}, acc1 = {};
    const size_t base = (size_t)b * 4096 * 1536 + (size_t)h * 64;

    for (int n0 = 0; n0 < 4096; n0 += 32) {
        const uint16_t* qp = qkv + base + (size_t)(n0 + nl) * 1536 + c0;
        const uint16_t* kp = qp + 512;
        const U8x16 qv = { *(const u16x8*)qp };
        const U8x16 kv = { *(const u16x8*)kp };
#pragma unroll
        for (int j = 0; j < 8; ++j) {
            const float qf = bf2f(qv.u[j]), kf = bf2f(kv.u[j]);
            pq[j] += qf * qf; pk[j] += kf * kf;
            qT[c0 + j][nl] = qv.u[j];
            kT[c0 + j][nl] = kv.u[j];
        }
        __syncthreads();
        {
            BfVec a, bb;
            load_frag(a,  &qT[(t0 >> 2) * 16 + row][half * 8]);
            load_frag(bb, &kT[(t0 & 3)  * 16 + row][half * 8]);
            acc0 = WMMA_BF16(a.v, bb.v, acc0);
        }
        {
            BfVec a, bb;
            load_frag(a,  &qT[(t1 >> 2) * 16 + row][half * 8]);
            load_frag(bb, &kT[(t1 & 3)  * 16 + row][half * 8]);
            acc1 = WMMA_BF16(a.v, bb.v, acc1);
        }
        __syncthreads();
    }

#pragma unroll
    for (int j = 0; j < 8; ++j) {
        atomicAdd(&ssq[c0 + j], pq[j]);
        atomicAdd(&ssk[c0 + j], pk[j]);
    }
    __syncthreads();

    const float tmp = temp[h];
    {
        const int c = (t0 >> 2) * 16 + half * 8, d = (t0 & 3) * 16 + row;
        const float invk = 1.f / fmaxf(sqrtf(ssk[d]), 1e-12f);
#pragma unroll
        for (int i = 0; i < 8; ++i)
            attn[c + i][d] = acc0[i] * (1.f / fmaxf(sqrtf(ssq[c + i]), 1e-12f)) * invk * tmp;
    }
    {
        const int c = (t1 >> 2) * 16 + half * 8, d = (t1 & 3) * 16 + row;
        const float invk = 1.f / fmaxf(sqrtf(ssk[d]), 1e-12f);
#pragma unroll
        for (int i = 0; i < 8; ++i)
            attn[c + i][d] = acc1[i] * (1.f / fmaxf(sqrtf(ssq[c + i]), 1e-12f)) * invk * tmp;
    }
    __syncthreads();

    if (tid < 64) {
        float m = -1e30f;
        for (int d = 0; d < 64; ++d) m = fmaxf(m, attn[tid][d]);
        float s = 0.f;
        for (int d = 0; d < 64; ++d) { float e = __expf(attn[tid][d] - m); attn[tid][d] = e; s += e; }
        const float inv = 1.f / s;
        uint16_t* ap = attn_out + ((size_t)bh * 64 + tid) * 64;
        for (int d = 0; d < 64; ++d) ap[d] = f2bf(attn[tid][d] * inv);
    }
}

// ============================================================
// K3: per (b,h): Out[n,c] = sum_d v[n,d]*attn[c,d]  (K=64)
// block = (bh, 128-token group); TDM stages V tile (128x64) and attn
// (64x64) into LDS, waves consume via ds_load_b128 fragments.
// grid = 64 bh * 32 groups = 2048 blocks
// ============================================================
__global__ __launch_bounds__(256) void k3_attn_v(
    const uint16_t* __restrict__ qkv, const uint16_t* __restrict__ attn,
    uint16_t* __restrict__ outb)
{
    __shared__ uint16_t vtile[128 * 64];   // token-major V slice (16 KB)
    __shared__ uint16_t atile[64 * 64];    // attn rows (c,d) (8 KB)

    const int bh  = blockIdx.x >> 5;       // 0..63
    const int ntg = blockIdx.x & 31;       // 0..31 (128 tokens each)
    const int b = bh >> 3, h = bh & 7;
    const int lane = threadIdx.x & 31;
    const int wave = threadIdx.x >> 5;
    const int row  = lane & 15;
    const int half = lane >> 4;

    if (wave == 0) {
        const uint16_t* vsrc = qkv + (size_t)b * 4096 * 1536 + 1024 + (size_t)h * 64
                                   + (size_t)(ntg * 128) * 1536;
        // V tile: 128 rows x 64 contiguous elems, row stride 1536 elems
        tdm_load_tile_2d((uint32_t)(uintptr_t)vtile, vsrc,
                         /*tile*/64, 128, /*tensor*/64, 4096, /*stride0*/1536);
        // attn block: 64x64 contiguous
        tdm_load_tile_2d((uint32_t)(uintptr_t)atile, attn + (size_t)bh * 4096,
                         /*tile*/64, 64, /*tensor*/64, 64, /*stride0*/64);
        __builtin_amdgcn_s_wait_tensorcnt(0);
    }
    __syncthreads();

    v8f acc[4] = {};
    const int tok = wave * 16 + row;       // token row inside LDS tile
#pragma unroll
    for (int kk = 0; kk < 64; kk += 32) {
        BfVec a;
        load_frag(a, &vtile[tok * 64 + kk + half * 8]);
#pragma unroll
        for (int ct = 0; ct < 4; ++ct) {
            BfVec bb;                      // B[d,c]: col c = attn row, d contiguous
            load_frag(bb, &atile[(ct * 16 + row) * 64 + kk + half * 8]);
            acc[ct] = WMMA_BF16(a.v, bb.v, acc[ct]);
        }
    }
    uint16_t* op = outb + ((size_t)b * 4096 + ntg * 128 + wave * 16) * 512 + h * 64;
#pragma unroll
    for (int ct = 0; ct < 4; ++ct)
#pragma unroll
        for (int i = 0; i < 8; ++i)
            op[(size_t)(half * 8 + i) * 512 + ct * 16 + row] = f2bf(acc[ct][i]);
}

// ============================================================
// K4: final[m,d] = sum_c out[m,c]*Wproj[d,c] + bproj[d]
// 2048 mt x 8 dg waves, register double-buffered
// ============================================================
__global__ __launch_bounds__(256) void k4_proj_gemm(
    const uint16_t* __restrict__ outb, const uint16_t* __restrict__ wpb,
    const float* __restrict__ bproj, float* __restrict__ out)
{
    const int lane = threadIdx.x & 31;
    const int gw   = blockIdx.x * 8 + (threadIdx.x >> 5);
    const int mt   = gw >> 3;
    const int dg   = gw & 7;
    const int row  = lane & 15;
    const int half = lane >> 4;

    const uint16_t* arow = outb + (size_t)(mt * 16 + row) * 512 + half * 8;
    const uint16_t* brow[4];
#pragma unroll
    for (int j = 0; j < 4; ++j)
        brow[j] = wpb + (size_t)(dg * 64 + j * 16 + row) * 512 + half * 8;

    v8f acc[4] = {};
    BfVec a0, a1, b0[4], b1[4];

    auto ldgrp = [&](BfVec& a, BfVec (&b)[4], int kk) {
        load_frag(a, arow + kk);
#pragma unroll
        for (int j = 0; j < 4; ++j) load_frag(b[j], brow[j] + kk);
    };
    auto mmgrp = [&](BfVec& a, BfVec (&b)[4]) {
#pragma unroll
        for (int j = 0; j < 4; ++j) acc[j] = WMMA_BF16(a.v, b[j].v, acc[j]);
    };

    ldgrp(a0, b0, 0);
#pragma unroll
    for (int kk = 0; kk < 512; kk += 64) {
        if (kk + 32 < 512) ldgrp(a1, b1, kk + 32);
        mmgrp(a0, b0);
        if (kk + 64 < 512) ldgrp(a0, b0, kk + 64);
        mmgrp(a1, b1);
    }
    float* op = out + (size_t)(mt * 16 + half * 8) * 512 + dg * 64;
#pragma unroll
    for (int j = 0; j < 4; ++j) {
        const float bias = bproj[dg * 64 + j * 16 + row];
#pragma unroll
        for (int i = 0; i < 8; ++i)
            op[(size_t)i * 512 + j * 16 + row] = acc[j][i] + bias;
    }
}

extern "C" void kernel_launch(void* const* d_in, const int* in_sizes, int n_in,
                              void* d_out, int out_size, void* d_ws, size_t ws_size,
                              hipStream_t stream) {
    const float* x     = (const float*)d_in[0];
    const float* Wqkv  = (const float*)d_in[1];
    const float* Wproj = (const float*)d_in[2];
    const float* bproj = (const float*)d_in[3];
    const float* temp  = (const float*)d_in[4];
    (void)in_sizes; (void)n_in; (void)out_size; (void)ws_size;

    char* ws = (char*)d_ws;
    uint16_t* qkv  = (uint16_t*)(ws);                       // 100,663,296 B
    uint16_t* outb = (uint16_t*)(ws + (size_t)100663296);   //  33,554,432 B
    uint16_t* attn = (uint16_t*)(ws + (size_t)134217728);   //     524,288 B
    uint16_t* xb   = (uint16_t*)(ws + (size_t)134742016);   //  33,554,432 B
    uint16_t* wqb  = (uint16_t*)(ws + (size_t)168296448);   //   1,572,864 B
    uint16_t* wpb  = (uint16_t*)(ws + (size_t)169869312);   //     524,288 B

    k0_cvt<<<8192, 256, 0, stream>>>(x,     xb,  16777216 / 8);
    k0_cvt<<< 384, 256, 0, stream>>>(Wqkv,  wqb,   786432 / 8);
    k0_cvt<<< 128, 256, 0, stream>>>(Wproj, wpb,   262144 / 8);

    k1_qkv_gemm <<<6144, 256, 0, stream>>>(xb, wqb, qkv);
    k2_chan_attn<<<  64, 256, 0, stream>>>(qkv, temp, attn);
    k3_attn_v   <<<2048, 256, 0, stream>>>(qkv, attn, outb);
    k4_proj_gemm<<<2048, 256, 0, stream>>>(outb, wpb, bproj, (float*)d_out);
}
